// SNNAnomalyDetector_40793599378015
// MI455X (gfx1250) — matrix operand order repaired and checked by reading, not compiled, not run on hardware
//
#include <hip/hip_runtime.h>
#include <hip/hip_fp16.h>

typedef _Float16 v16h __attribute__((ext_vector_type(16)));
typedef float    v8f  __attribute__((ext_vector_type(8)));

#define T_STEPS 160
#define BATCH   2048
#define IN_DIM  400
#define H1      256
#define H2      128
#define NCLS    7
#define KP      416      // IN padded to multiple of 32
#define WSTRA   424      // k-stride (halves) for spike staging tile (bank spread)
#define S1_STR  264      // k-stride for s1 spike tiles
#define S2_STR  136      // k-stride for s2 spike tiles

// ---------------- workspace layout (bytes) ----------------
static constexpr size_t WIN1F_OFF  = 0;                                // f16 [256][416]
static constexpr size_t WREC1F_OFF = WIN1F_OFF  + (size_t)H1 * KP * 2; // f16 [256][256]
static constexpr size_t WIN2F_OFF  = WREC1F_OFF + (size_t)H1 * H1 * 2; // f16 [128][256] (BN-folded)
static constexpr size_t WREC2F_OFF = WIN2F_OFF  + (size_t)H2 * H1 * 2; // f16 [128][128]
static constexpr size_t BIAS2F_OFF = WREC2F_OFF + (size_t)H2 * H2 * 2; // f32 [128] (BN-folded bias)
static constexpr size_t AVG_OFF    = BIAS2F_OFF + (size_t)H2 * 4;      // f32 [2048][128]
static constexpr size_t INP1_OFF   = AVG_OFF + (size_t)BATCH * H2 * 4; // f16 [160*2048][256]

// ---------------- WMMA helpers ----------------
__device__ __forceinline__ v8f wmma16(v16h a, v16h b, v8f c) {
  // D = A(16x32 f16) * B(32x16 f16) + C(16x16 f32)
  return __builtin_amdgcn_wmma_f32_16x16x32_f16(false, a, false, b, (short)0, c,
                                                false, false);
}

// A fragment (16x32, f16), row-major source with row stride `strideh` halves.
// ISA layout: lanes 0-15 m=lane, halves 0-7 -> k0..k0+7, halves 8-15 -> k0+16..23;
// lanes 16-31 shifted by +8 in k.  Two 16-byte loads per lane.
__device__ __forceinline__ v16h load_fragA(const _Float16* base, int strideh,
                                           int k0, int lane) {
  const int m = lane & 15, khi = lane >> 4;
  const _Float16* p = base + (size_t)m * strideh + k0 + 8 * khi;
  union { v16h v; uint4 q[2]; } u;
  u.q[0] = *reinterpret_cast<const uint4*>(p);
  u.q[1] = *reinterpret_cast<const uint4*>(p + 16);
  return u.v;
}

// B fragment (32x16, f16) from an N-major weight matrix W[n][k] (stride `strideh`).
// Layout matches A: lanes 0-15 n=lane, halves 0-7 -> k0..k0+7, halves 8-15 -> k0+16..23; lanes 16-31 +8 in k.
__device__ __forceinline__ v16h load_fragB(const _Float16* base, int strideh,
                                           int k0, int lane) {
  const int n = lane & 15, kh = lane >> 4;
  const _Float16* p = base + (size_t)n * strideh + k0 + 8 * kh;
  union { v16h v; uint4 q[2]; } u;
  u.q[0] = *reinterpret_cast<const uint4*>(p);
  u.q[1] = *reinterpret_cast<const uint4*>(p + 16);
  return u.v;
}

// ---------------- kernel 0: fold BN into weights, convert to f16 ----------------
__global__ void __launch_bounds__(256)
snn_prep(const float* __restrict__ W_in1, const float* __restrict__ W_rec1,
         const float* __restrict__ W_in2, const float* __restrict__ b_in2,
         const float* __restrict__ W_rec2,
         const float* __restrict__ g1, const float* __restrict__ be1,
         const float* __restrict__ rm1, const float* __restrict__ rv1,
         _Float16* __restrict__ win1f, _Float16* __restrict__ wrec1f,
         _Float16* __restrict__ win2f, _Float16* __restrict__ wrec2f,
         float* __restrict__ bias2f) {
  const int N0 = H1 * KP, N1 = H1 * H1, N2 = H2 * H1, N3 = H2 * H2, N4 = H2;
  const int total = N0 + N1 + N2 + N3 + N4;
  for (int i = blockIdx.x * blockDim.x + threadIdx.x; i < total;
       i += gridDim.x * blockDim.x) {
    int j = i;
    if (j < N0) {  // W_in1 -> f16, zero-pad k 400..415
      const int k = j % KP, n = j / KP;
      win1f[j] = (k < IN_DIM) ? (_Float16)W_in1[n * IN_DIM + k] : (_Float16)0.0f;
      continue;
    }
    j -= N0;
    if (j < N1) { wrec1f[j] = (_Float16)W_rec1[j]; continue; }
    j -= N1;
    if (j < N2) {  // fold BN1 scale into W_in2 columns
      const int k = j % H1;
      const float sc = g1[k] * rsqrtf(rv1[k] + 1e-5f);
      win2f[j] = (_Float16)(W_in2[j] * sc);
      continue;
    }
    j -= N2;
    if (j < N3) { wrec2f[j] = (_Float16)W_rec2[j]; continue; }
    j -= N3;
    {  // bias2f[j] = b_in2[j] + sum_k shift1[k] * W_in2[j][k]
      float s = b_in2[j];
      for (int k = 0; k < H1; ++k) {
        const float sc = g1[k] * rsqrtf(rv1[k] + 1e-5f);
        s += (be1[k] - rm1[k] * sc) * W_in2[j * H1 + k];
      }
      bias2f[j] = s;
    }
  }
}

// ---------------- kernel A: time-parallel input projection ----------------
// inp1[t*B+b][n] = sum_k spikes[t,b,k] * W_in1[n,k] + b_in1[n]   (f16 out)
// Block = 8 waves; block tile M=64 x N=256.  Each wave owns a 32-column slice
// and register-caches ALL its weight B-fragments (13 k-chunks x 2 n-tiles,
// 208 VGPRs) once, loaded straight from global.  Per M-tile only A fragments
// are read from LDS (1 ds_load_b128 per WMMA), software-pipelined one
// k-chunk ahead so LDS latency overlaps the WMMAs.
__global__ void __launch_bounds__(256, 1)
snn_input_gemm(const float* __restrict__ x, const float* __restrict__ b_in1,
               const _Float16* __restrict__ win1f, _Float16* __restrict__ inp1) {
  __shared__ _Float16 sX[64 * WSTRA];   // 54,272 B spike staging tile
  const int tid = threadIdx.x;
  const int lane = tid & 31;
  const int w = tid >> 5;               // wave 0..7 -> columns [32w, 32w+32)
  const int nlo = lane & 15;
  const int mh = (lane >> 4) * 8;       // C layout: lanes 16-31 hold M = r+8

  // Preload weight fragments for this wave's two 16-col tiles (all 13 k-chunks).
  v16h Bw[2][13];
  #pragma unroll
  for (int nt = 0; nt < 2; ++nt) {
    #pragma unroll
    for (int kc = 0; kc < 13; ++kc)
      Bw[nt][kc] = load_fragB(win1f + (size_t)(32 * w + 16 * nt) * KP, KP,
                              kc * 32, lane);
  }
  const float bias0 = b_in1[32 * w + nlo];
  const float bias1 = b_in1[32 * w + 16 + nlo];

  const int ntiles = (T_STEPS * BATCH) / 64;  // 5120
  for (int tile = blockIdx.x; tile < ntiles; tile += gridDim.x) {
    const size_t row0 = (size_t)tile * 64;
    // Stage spike tile (64x400 f32) -> f16 LDS, pad k 400..415 with 0.
    for (int i = tid; i < 64 * (KP / 4); i += 256) {
      const int r = i / (KP / 4), c = (i % (KP / 4)) * 4;
      _Float16* dst = &sX[r * WSTRA + c];
      if (c < IN_DIM) {
        const float4 v =
            *reinterpret_cast<const float4*>(&x[(row0 + r) * IN_DIM + c]);
        dst[0] = (_Float16)v.x; dst[1] = (_Float16)v.y;
        dst[2] = (_Float16)v.z; dst[3] = (_Float16)v.w;
      } else {
        dst[0] = (_Float16)0.f; dst[1] = (_Float16)0.f;
        dst[2] = (_Float16)0.f; dst[3] = (_Float16)0.f;
      }
    }
    // Prefetch next tile's spike rows (emits global_prefetch_b8).
    if (tile + (int)gridDim.x < ntiles) {
      const char* nx =
          (const char*)&x[(row0 + (size_t)gridDim.x * 64) * IN_DIM];
      __builtin_prefetch(nx + tid * 512, 0, 1);
    }
    __syncthreads();

    #pragma unroll
    for (int mg = 0; mg < 4; ++mg) {   // four 16-row groups
      v8f c0, c1;
      #pragma unroll
      for (int r = 0; r < 8; ++r) { c0[r] = 0.f; c1[r] = 0.f; }
      const _Float16* xb = &sX[(16 * mg) * WSTRA];
      // software-pipelined k loop: load chunk kc+1 while multiplying chunk kc
      v16h a = load_fragA(xb, WSTRA, 0, lane);
      #pragma unroll
      for (int kc = 0; kc < KP / 32 - 1; ++kc) {
        const v16h an = load_fragA(xb, WSTRA, (kc + 1) * 32, lane);
        c0 = wmma16(a, Bw[0][kc], c0);
        c1 = wmma16(a, Bw[1][kc], c1);
        a = an;
      }
      c0 = wmma16(a, Bw[0][KP / 32 - 1], c0);
      c1 = wmma16(a, Bw[1][KP / 32 - 1], c1);
      // store this M-group immediately (keeps only 2 accumulators live)
      #pragma unroll
      for (int r = 0; r < 8; ++r) {
        const size_t m = row0 + 16 * mg + r + mh;
        inp1[m * H1 + 32 * w + nlo]      = (_Float16)(c0[r] + bias0);
        inp1[m * H1 + 32 * w + 16 + nlo] = (_Float16)(c1[r] + bias1);
      }
    }
    __syncthreads();
  }
}

// ---------------- kernel B: sequential recurrence (per-row independent) ----------------
// Block = 8 waves, owns 16 batch rows for all 160 steps.  Weight B-fragments live in
// VGPRs for the whole loop; spike tiles ping-pong through LDS; 2 barriers/step.
// A-fragment loads are software-pipelined one k-chunk ahead to keep LDS latency
// off the serial critical path.
__global__ void __launch_bounds__(256, 1)
snn_recurrent(const _Float16* __restrict__ inp1,
              const _Float16* __restrict__ wrec1f,
              const _Float16* __restrict__ win2f,
              const _Float16* __restrict__ wrec2f,
              const float* __restrict__ bias2f,
              float* __restrict__ avgspk) {
  __shared__ _Float16 s1[2][16 * S1_STR];
  __shared__ _Float16 s2[2][16 * S2_STR];
  const int tid = threadIdx.x, lane = tid & 31, w = tid >> 5;  // w = 0..7
  const int r0 = blockIdx.x * 16;
  const int nlo = lane & 15;
  const int mh = (lane >> 4) * 8;

  // zero the spike ping-pong buffers (initial carry is zero)
  for (int i = tid; i < 2 * 16 * S1_STR; i += 256) (&s1[0][0])[i] = (_Float16)0.f;
  for (int i = tid; i < 2 * 16 * S2_STR; i += 256) (&s2[0][0])[i] = (_Float16)0.f;

  // Preload all weight B-fragments into registers (constant across time).
  v16h Brec1[2][8], Bin2[8], Brec2[4];
  #pragma unroll
  for (int nt = 0; nt < 2; ++nt) {
    #pragma unroll
    for (int kc = 0; kc < 8; ++kc)
      Brec1[nt][kc] =
          load_fragB(wrec1f + (size_t)(32 * w + 16 * nt) * H1, H1, kc * 32, lane);
  }
  #pragma unroll
  for (int kc = 0; kc < 8; ++kc)
    Bin2[kc] = load_fragB(win2f + (size_t)(16 * w) * H1, H1, kc * 32, lane);
  #pragma unroll
  for (int kc = 0; kc < 4; ++kc)
    Brec2[kc] = load_fragB(wrec2f + (size_t)(16 * w) * H2, H2, kc * 32, lane);

  const float bias2 = bias2f[16 * w + nlo];

  v8f acc1[2], acc2, accT;
  #pragma unroll
  for (int r = 0; r < 8; ++r) {
    acc1[0][r] = 0.f; acc1[1][r] = 0.f; acc2[r] = 0.f; accT[r] = 0.f;
  }
  __syncthreads();

  for (int t = 0; t < T_STEPS; ++t) {
    const int p = t & 1, q = p ^ 1;

    // ---- layer 1: mem1 = 0.8*mem1 + inp1[t] + s1_prev @ W_rec1^T ----
    const size_t ibase = ((size_t)t * BATCH + r0) * H1;
    v8f c0, c1;
    #pragma unroll
    for (int r = 0; r < 8; ++r) {
      const int m = r + mh;
      c0[r] = acc1[0][r] * 0.8f +
              (float)inp1[ibase + (size_t)m * H1 + (32 * w + nlo)];
      c1[r] = acc1[1][r] * 0.8f +
              (float)inp1[ibase + (size_t)m * H1 + (32 * w + 16 + nlo)];
    }
    {
      v16h a = load_fragA(&s1[p][0], S1_STR, 0, lane);
      #pragma unroll
      for (int kc = 0; kc < 7; ++kc) {
        const v16h an = load_fragA(&s1[p][0], S1_STR, (kc + 1) * 32, lane);
        c0 = wmma16(a, Brec1[0][kc], c0);
        c1 = wmma16(a, Brec1[1][kc], c1);
        a = an;
      }
      c0 = wmma16(a, Brec1[0][7], c0);
      c1 = wmma16(a, Brec1[1][7], c1);
    }
    acc1[0] = c0; acc1[1] = c1;
    #pragma unroll
    for (int r = 0; r < 8; ++r) {  // spike (sigmoid(x)>0.5 <=> x>0)
      const int m = r + mh;
      s1[q][m * S1_STR + 32 * w + nlo]      = (_Float16)(c0[r] > 0.f ? 1.f : 0.f);
      s1[q][m * S1_STR + 32 * w + 16 + nlo] = (_Float16)(c1[r] > 0.f ? 1.f : 0.f);
    }
    __syncthreads();

    // ---- layer 2: mem2 = 0.8*mem2 + bias2' + s1 @ Win2'^T + s2_prev @ Wrec2^T ----
    v8f c2;
    #pragma unroll
    for (int r = 0; r < 8; ++r) c2[r] = acc2[r] * 0.8f + bias2;
    {
      // pipelined across both GEMMs: s1 @ Win2' (8 chunks) then s2 @ Wrec2 (4)
      v16h a = load_fragA(&s1[q][0], S1_STR, 0, lane);
      #pragma unroll
      for (int kc = 0; kc < 8; ++kc) {
        const v16h an = (kc < 7)
            ? load_fragA(&s1[q][0], S1_STR, (kc + 1) * 32, lane)
            : load_fragA(&s2[p][0], S2_STR, 0, lane);
        c2 = wmma16(a, Bin2[kc], c2);
        a = an;
      }
      #pragma unroll
      for (int kc = 0; kc < 3; ++kc) {
        const v16h an = load_fragA(&s2[p][0], S2_STR, (kc + 1) * 32, lane);
        c2 = wmma16(a, Brec2[kc], c2);
        a = an;
      }
      c2 = wmma16(a, Brec2[3], c2);
    }
    acc2 = c2;
    #pragma unroll
    for (int r = 0; r < 8; ++r) {
      const int m = r + mh;
      const float sp = c2[r] > 0.f ? 1.f : 0.f;
      s2[q][m * S2_STR + 16 * w + nlo] = (_Float16)sp;
      if (t >= T_STEPS - 10) accT[r] += sp;   // uniform branch
    }
    __syncthreads();
  }

  // mean spike rate over last 10 steps
  #pragma unroll
  for (int r = 0; r < 8; ++r) {
    const int m = r + mh;
    avgspk[(size_t)(r0 + m) * H2 + 16 * w + nlo] = accT[r] * 0.1f;
  }
}

// ---------------- kernel C: BN2 + readout + softmax ----------------
__global__ void __launch_bounds__(256)
snn_readout(const float* __restrict__ avgspk,
            const float* __restrict__ g2, const float* __restrict__ be2,
            const float* __restrict__ rm2, const float* __restrict__ rv2,
            const float* __restrict__ W_ro, const float* __restrict__ b_ro,
            float* __restrict__ out) {
  __shared__ float sw[NCLS * H2];
  __shared__ float ssc[H2], ssh[H2];
  const int tid = threadIdx.x;
  for (int i = tid; i < NCLS * H2; i += 256) sw[i] = W_ro[i];
  for (int i = tid; i < H2; i += 256) {
    const float sc = g2[i] * rsqrtf(rv2[i] + 1e-5f);
    ssc[i] = sc;
    ssh[i] = be2[i] - rm2[i] * sc;
  }
  __syncthreads();

  const int b = blockIdx.x * 256 + tid;
  float lg[NCLS];
  #pragma unroll
  for (int c = 0; c < NCLS; ++c) lg[c] = b_ro[c];
  for (int j = 0; j < H2; ++j) {
    const float z = avgspk[(size_t)b * H2 + j] * ssc[j] + ssh[j];
    #pragma unroll
    for (int c = 0; c < NCLS; ++c) lg[c] += z * sw[c * H2 + j];
  }
  float mx = lg[0];
  #pragma unroll
  for (int c = 1; c < NCLS; ++c) mx = fmaxf(mx, lg[c]);
  float s = 0.f, e0 = 0.f;
  #pragma unroll
  for (int c = 0; c < NCLS; ++c) {
    const float e = expf(lg[c] - mx);
    if (c == 0) e0 = e;
    s += e;
  }
  #pragma unroll
  for (int c = 0; c < NCLS; ++c) out[(size_t)b * NCLS + c] = lg[c];
  out[(size_t)BATCH * NCLS + b] = 1.f - e0 / s;
}

// ---------------- launch ----------------
extern "C" void kernel_launch(void* const* d_in, const int* in_sizes, int n_in,
                              void* d_out, int out_size, void* d_ws, size_t ws_size,
                              hipStream_t stream) {
  (void)in_sizes; (void)n_in; (void)out_size; (void)ws_size;
  const float* x      = (const float*)d_in[0];
  const float* W_in1  = (const float*)d_in[1];
  const float* b_in1  = (const float*)d_in[2];
  const float* W_rec1 = (const float*)d_in[3];
  const float* g1  = (const float*)d_in[4];
  const float* be1 = (const float*)d_in[5];
  const float* rm1 = (const float*)d_in[6];
  const float* rv1 = (const float*)d_in[7];
  const float* W_in2  = (const float*)d_in[8];
  const float* b_in2  = (const float*)d_in[9];
  const float* W_rec2 = (const float*)d_in[10];
  const float* g2  = (const float*)d_in[11];
  const float* be2 = (const float*)d_in[12];
  const float* rm2 = (const float*)d_in[13];
  const float* rv2 = (const float*)d_in[14];
  const float* W_ro = (const float*)d_in[15];
  const float* b_ro = (const float*)d_in[16];

  char* ws = (char*)d_ws;  // requires ws_size >= ~170 MB (inp1 staging is the bulk)
  _Float16* win1f  = (_Float16*)(ws + WIN1F_OFF);
  _Float16* wrec1f = (_Float16*)(ws + WREC1F_OFF);
  _Float16* win2f  = (_Float16*)(ws + WIN2F_OFF);
  _Float16* wrec2f = (_Float16*)(ws + WREC2F_OFF);
  float*    bias2f = (float*)(ws + BIAS2F_OFF);
  float*    avgspk = (float*)(ws + AVG_OFF);
  _Float16* inp1   = (_Float16*)(ws + INP1_OFF);

  snn_prep<<<dim3(512), dim3(256), 0, stream>>>(
      W_in1, W_rec1, W_in2, b_in2, W_rec2, g1, be1, rm1, rv1,
      win1f, wrec1f, win2f, wrec2f, bias2f);

  snn_input_gemm<<<dim3(512), dim3(256), 0, stream>>>(x, b_in1, win1f, inp1);

  snn_recurrent<<<dim3(BATCH / 16), dim3(256), 0, stream>>>(
      inp1, wrec1f, win2f, wrec2f, bias2f, avgspk);

  snn_readout<<<dim3(BATCH / 256), dim3(256), 0, stream>>>(
      avgspk, g2, be2, rm2, rv2, W_ro, b_ro, (float*)d_out);
}
